// NNNet_46342697124055
// MI455X (gfx1250) — compile-verified, matrix operand order checked by reading, not run on hardware
//
#include <hip/hip_runtime.h>

#define N_NODES 10000
#define N_EDGES 50000
#define F_INN   40
#define F_E     10
#define L_DIM   64
#define NCHUNK  130          // K = 4096 (outer product) + 64 (bias rows), step 32
#define MROWS   64           // edges per block (4 sub-tiles of 16)
#define NBLK    ((N_EDGES + MROWS - 1) / MROWS)   // 782

typedef __attribute__((ext_vector_type(16))) _Float16 v16h;
typedef __attribute__((ext_vector_type(8)))  float    v8f;

union V16U { v16h v; uint4 q[2]; };

// ---------------------------------------------------------------------------
// Small prep kernels
// ---------------------------------------------------------------------------
__global__ void deg_kernel(const int* __restrict__ eidx, float* __restrict__ deg) {
    int e = blockIdx.x * blockDim.x + threadIdx.x;
    if (e < N_EDGES) atomicAdd(&deg[eidx[N_EDGES + e]], 1.0f);   // dst row
}

__global__ void invdeg_kernel(float* __restrict__ deg) {
    int n = blockIdx.x * blockDim.x + threadIdx.x;
    if (n < N_NODES) deg[n] = 1.0f / fmaxf(deg[n], 1.0f);
}

// h1 = relu(edge_attr @ nn1_w + nn1_b), stored f16 [E, 64]
__global__ void h1_kernel(const float* __restrict__ ea, const float* __restrict__ w,
                          const float* __restrict__ b, _Float16* __restrict__ h1) {
    int idx = blockIdx.x * blockDim.x + threadIdx.x;
    if (idx >= N_EDGES * L_DIM) return;
    int e = idx >> 6, o = idx & 63;
    float acc = b[o];
    #pragma unroll
    for (int k = 0; k < F_E; ++k) acc += ea[e * F_E + k] * w[k * L_DIM + o];
    h1[idx] = (_Float16)fmaxf(acc, 0.0f);
}

// Pre-swizzle B = reshape(nn2_w,[4096,64]) (+ bias rows) into WMMA B-fragment
// order: halves [((t*4+w)*32+lane)*16 + j], j<8 -> K=8h+j, j>=8 -> K=8h+8+j.
__global__ void prepB_kernel(const float* __restrict__ nn2w,
                             const float* __restrict__ nn2b,
                             _Float16* __restrict__ Bsw) {
    int gid = blockIdx.x * blockDim.x + threadIdx.x;   // NCHUNK * 128 threads
    int t = gid >> 7;
    int rem = gid & 127;
    int lane = rem & 31;
    int c = (rem >> 5) * 16 + (lane & 15);             // output column 0..63
    int basep = (lane >> 4) * 8;
    _Float16* outp = Bsw + (size_t)gid * 16;
    #pragma unroll
    for (int j = 0; j < 16; ++j) {
        int pat = basep + j + ((j >= 8) ? 8 : 0);      // K offset within 32-chunk
        float v;
        if (t < 128) {
            int kg = (t >> 1) * 64 + (t & 1) * 32 + pat;   // k_h*64 + i
            v = nn2w[(size_t)kg * 64 + c];
        } else {
            int i = (t - 128) * 32 + pat;
            v = nn2b[i * 64 + c];
        }
        outp[j] = (_Float16)v;
    }
}

// out = relu(x @ lin0_w + lin0_b); init h = out; keep f32 + f16 copies
__global__ void lin0_kernel(const float* __restrict__ x, const float* __restrict__ w,
                            const float* __restrict__ b, float* __restrict__ out32,
                            _Float16* __restrict__ out16, float* __restrict__ h) {
    int idx = blockIdx.x * blockDim.x + threadIdx.x;
    if (idx >= N_NODES * L_DIM) return;
    int n = idx >> 6, o = idx & 63;
    float acc = b[o];
    #pragma unroll
    for (int k = 0; k < F_INN; ++k) acc += x[n * F_INN + k] * w[k * L_DIM + o];
    float v = fmaxf(acc, 0.0f);
    out32[idx] = v; out16[idx] = (_Float16)v; h[idx] = v;
}

// ---------------------------------------------------------------------------
// Main kernel: fused edge-net + per-edge matvec as one WMMA GEMM.
//   msg[64 x 64] = Z[64 x 4160] @ B[4160 x 64] per block, Z built on the fly:
//   Z[e, kh*64+i] = h1[e,kh] * out_src[e,i]  (+ identity rows for nn2_b).
// 4 waves/block; wave w owns cols [16w,16w+16) over 4 edge sub-tiles (B
// fragments reused 4x). Per chunk, ALL 8 scaled A-fragments are materialized
// in distinct registers before the 8 back-to-back WMMAs, so every WMMA's
// WAR hazard window is covered by >=7 real instructions (no v_nops) and the
// scheduler can interleave freely. h1 LDS carries an extra column fixed to
// 1.0 so the nn2_b bias chunks are just kh==64 of the same uniform loop.
// Padding edges get zeroed A rows -> contribute exactly 0.
// ---------------------------------------------------------------------------
__global__ __launch_bounds__(128)
void edge_msg_kernel(const int* __restrict__ eidx,
                     const _Float16* __restrict__ out16,
                     const _Float16* __restrict__ h1,
                     const _Float16* __restrict__ Bsw,
                     float* __restrict__ agg) {
    __shared__ _Float16 outs[MROWS * 72];   // padded rows -> no bank conflicts
    __shared__ _Float16 h1s[MROWS * 72];    // col 64 = 1.0 (bias pass-through)
    __shared__ int dsts[MROWS];

    const int tid = threadIdx.x;
    const int ebase = blockIdx.x * MROWS;

    // gather MROWS rows of out16[src] and h1 into LDS, 8 halves per slot
    #pragma unroll
    for (int i = 0; i < 4; ++i) {
        int slot = tid + i * 128;              // 0..511
        int row = slot >> 3, part = slot & 7;  // row 0..63
        int e = ebase + row;
        if (e < N_EDGES) {
            int sn = eidx[e];
            *(uint4*)(&outs[row * 72 + part * 8]) =
                *((const uint4*)(out16 + (size_t)sn * L_DIM) + part);
            *(uint4*)(&h1s[row * 72 + part * 8]) =
                *((const uint4*)(h1 + (size_t)e * L_DIM) + part);
        } else {
            uint4 z = {0u, 0u, 0u, 0u};
            *(uint4*)(&outs[row * 72 + part * 8]) = z;
            *(uint4*)(&h1s[row * 72 + part * 8]) = z;
        }
    }
    if (tid < MROWS) {
        dsts[tid] = (ebase + tid < N_EDGES) ? eidx[N_EDGES + ebase + tid] : 0;
        h1s[tid * 72 + 64] = (_Float16)1.0f;   // scale for bias chunks
    }
    __syncthreads();

    const int wv   = tid >> 5;          // column tile 0..3
    const int lane = tid & 31;
    const int r    = lane & 15;         // A row-in-subtile / C col
    const int hs   = lane >> 4;
    const int base = hs * 8;

    // per-lane A slices: i in {base..base+7, base+16..base+23} (+32 for a1)
    V16U a0[4], a1[4];
    #pragma unroll
    for (int s4 = 0; s4 < 4; ++s4) {
        int rr = s4 * 16 + r;
        a0[s4].q[0] = *(const uint4*)&outs[rr * 72 + base];
        a0[s4].q[1] = *(const uint4*)&outs[rr * 72 + base + 16];
        a1[s4].q[0] = *(const uint4*)&outs[rr * 72 + base + 32];
        a1[s4].q[1] = *(const uint4*)&outs[rr * 72 + base + 48];
    }

    const size_t cs = (size_t)4 * 32 * 16;                 // chunk stride (halves)
    const _Float16* Bl = Bsw + (size_t)(wv * 32 + lane) * 16;

#define WMMA(ACC, A, B) \
    ACC = __builtin_amdgcn_wmma_f32_16x16x32_f16(false, A, false, B, (short)0, \
                                                 ACC, false, false)

    // preload B chunk pair 0
    V16U b0, b1;
    b0.q[0] = *(const uint4*)(Bl);
    b0.q[1] = *(const uint4*)(Bl + 8);
    b1.q[0] = *(const uint4*)(Bl + cs);
    b1.q[1] = *(const uint4*)(Bl + cs + 8);

    v8f acc[4] = {};
    // kh == 64 is the bias pass (chunks 128/129, scale column == 1.0)
    #pragma unroll 1
    for (int kh = 0; kh <= 64; ++kh) {
        __builtin_prefetch(Bl + (size_t)(2 * kh + 8) * cs, 0, 1);
        _Float16 s0 = h1s[(0 * 16 + r) * 72 + kh];
        _Float16 s1 = h1s[(1 * 16 + r) * 72 + kh];
        _Float16 s2 = h1s[(2 * 16 + r) * 72 + kh];
        _Float16 s3 = h1s[(3 * 16 + r) * 72 + kh];
        // all 8 fragments live at the first WMMA -> 8 distinct register sets
        v16h F0 = a0[0].v * s0;
        v16h F1 = a1[0].v * s0;
        v16h F2 = a0[1].v * s1;
        v16h F3 = a1[1].v * s1;
        v16h F4 = a0[2].v * s2;
        v16h F5 = a1[2].v * s2;
        v16h F6 = a0[3].v * s3;
        v16h F7 = a1[3].v * s3;
        WMMA(acc[0], F0, b0.v);
        WMMA(acc[0], F1, b1.v);
        WMMA(acc[1], F2, b0.v);
        WMMA(acc[1], F3, b1.v);
        WMMA(acc[2], F4, b0.v);
        WMMA(acc[2], F5, b1.v);
        WMMA(acc[3], F6, b0.v);
        WMMA(acc[3], F7, b1.v);
        // load next chunk pair (Bsw has 2 slack chunks for the final lookahead)
        const _Float16* pn = Bl + (size_t)(2 * (kh + 1)) * cs;
        b0.q[0] = *(const uint4*)(pn);
        b0.q[1] = *(const uint4*)(pn + 8);
        b1.q[0] = *(const uint4*)(pn + cs);
        b1.q[1] = *(const uint4*)(pn + cs + 8);
    }
#undef WMMA

    // C layout: lane%16 = col, VGPR v -> edge row v + 8*(lane/16)
    const int c0 = wv * 16;
    #pragma unroll
    for (int s4 = 0; s4 < 4; ++s4) {
        #pragma unroll
        for (int v = 0; v < 8; ++v) {
            int er = s4 * 16 + v + 8 * hs;
            atomicAdd(&agg[(size_t)dsts[er] * L_DIM + c0 + r], acc[s4][v]);
        }
    }
}

// ---------------------------------------------------------------------------
// Node update: m = relu(agg*inv_deg + out@conv_root + bias); GRU step
// ---------------------------------------------------------------------------
__global__ __launch_bounds__(64)
void node_update_kernel(const float* __restrict__ agg, const float* __restrict__ invdeg,
                        const float* __restrict__ croot, const float* __restrict__ cbias,
                        const float* __restrict__ wih, const float* __restrict__ whh,
                        const float* __restrict__ bih, const float* __restrict__ bhh,
                        float* __restrict__ out32, _Float16* __restrict__ out16,
                        float* __restrict__ h) {
    int n = blockIdx.x, o = threadIdx.x;
    __shared__ float os[64], hsh[64], ms[64];
    os[o] = out32[n * 64 + o];
    hsh[o] = h[n * 64 + o];
    __syncthreads();
    float cr = cbias[o];
    #pragma unroll 8
    for (int k = 0; k < 64; ++k) cr += os[k] * croot[k * 64 + o];
    float m = fmaxf(agg[n * 64 + o] * invdeg[n] + cr, 0.0f);
    ms[o] = m;
    __syncthreads();
    float gi0 = bih[o], gi1 = bih[64 + o], gi2 = bih[128 + o];
    float gh0 = bhh[o], gh1 = bhh[64 + o], gh2 = bhh[128 + o];
    #pragma unroll 4
    for (int k = 0; k < 64; ++k) {
        float mk = ms[k], hk = hsh[k];
        gi0 += mk * wih[k * 192 + o];
        gi1 += mk * wih[k * 192 + 64 + o];
        gi2 += mk * wih[k * 192 + 128 + o];
        gh0 += hk * whh[k * 192 + o];
        gh1 += hk * whh[k * 192 + 64 + o];
        gh2 += hk * whh[k * 192 + 128 + o];
    }
    float rr = 1.0f / (1.0f + expf(-(gi0 + gh0)));
    float zz = 1.0f / (1.0f + expf(-(gi1 + gh1)));
    float ng = tanhf(gi2 + rr * gh2);
    float hn = (1.0f - zz) * ng + zz * hsh[o];
    h[n * 64 + o] = hn;
    out32[n * 64 + o] = hn;
    out16[n * 64 + o] = (_Float16)hn;
}

// readout: y = relu(h @ lin1_w + lin1_b) @ lin2_w + lin2_b
__global__ __launch_bounds__(64)
void final_kernel(const float* __restrict__ h, const float* __restrict__ w1,
                  const float* __restrict__ b1, const float* __restrict__ w2,
                  const float* __restrict__ b2, float* __restrict__ out) {
    int n = blockIdx.x, o = threadIdx.x;
    __shared__ float hsh[64], ts[64];
    hsh[o] = h[n * 64 + o];
    __syncthreads();
    float acc = b1[o];
    #pragma unroll 8
    for (int k = 0; k < 64; ++k) acc += hsh[k] * w1[k * 64 + o];
    ts[o] = fmaxf(acc, 0.0f) * w2[o];
    __syncthreads();
    for (int s = 32; s > 0; s >>= 1) {
        if (o < s) ts[o] += ts[o + s];
        __syncthreads();
    }
    if (o == 0) out[n] = ts[0] + b2[0];
}

// ---------------------------------------------------------------------------
extern "C" void kernel_launch(void* const* d_in, const int* in_sizes, int n_in,
                              void* d_out, int out_size, void* d_ws, size_t ws_size,
                              hipStream_t stream) {
    const float* x     = (const float*)d_in[0];
    const int*   eidx  = (const int*)d_in[1];
    const float* ea    = (const float*)d_in[2];
    const float* l0w   = (const float*)d_in[3];
    const float* l0b   = (const float*)d_in[4];
    const float* nn1w  = (const float*)d_in[5];
    const float* nn1b  = (const float*)d_in[6];
    const float* nn2w  = (const float*)d_in[7];
    const float* nn2b  = (const float*)d_in[8];
    const float* croot = (const float*)d_in[9];
    const float* cbias = (const float*)d_in[10];
    const float* wih   = (const float*)d_in[11];
    const float* whh   = (const float*)d_in[12];
    const float* bih   = (const float*)d_in[13];
    const float* bhh   = (const float*)d_in[14];
    const float* l1w   = (const float*)d_in[15];
    const float* l1b   = (const float*)d_in[16];
    const float* l2w   = (const float*)d_in[17];
    const float* l2b   = (const float*)d_in[18];
    float* yout = (float*)d_out;

    char* ws = (char*)d_ws;
    size_t off = 0;
    auto alloc = [&](size_t bytes) -> void* {
        void* p = ws + off;
        off = (off + bytes + 255) & ~(size_t)255;
        return p;
    };
    float*    invdeg = (float*)alloc((size_t)N_NODES * 4);
    _Float16* h1     = (_Float16*)alloc((size_t)N_EDGES * L_DIM * 2);
    _Float16* Bsw    = (_Float16*)alloc((size_t)(NCHUNK + 2) * 4 * 32 * 16 * 2);
    float*    out32  = (float*)alloc((size_t)N_NODES * L_DIM * 4);
    _Float16* out16  = (_Float16*)alloc((size_t)N_NODES * L_DIM * 2);
    float*    hbuf   = (float*)alloc((size_t)N_NODES * L_DIM * 4);
    float*    agg    = (float*)alloc((size_t)N_NODES * L_DIM * 4);

    hipMemsetAsync(invdeg, 0, (size_t)N_NODES * 4, stream);
    deg_kernel<<<(N_EDGES + 255) / 256, 256, 0, stream>>>(eidx, invdeg);
    invdeg_kernel<<<(N_NODES + 255) / 256, 256, 0, stream>>>(invdeg);
    h1_kernel<<<(N_EDGES * L_DIM + 255) / 256, 256, 0, stream>>>(ea, nn1w, nn1b, h1);
    prepB_kernel<<<NCHUNK, 128, 0, stream>>>(nn2w, nn2b, Bsw);
    lin0_kernel<<<(N_NODES * L_DIM + 255) / 256, 256, 0, stream>>>(x, l0w, l0b,
                                                                   out32, out16, hbuf);
    for (int it = 0; it < 3; ++it) {
        hipMemsetAsync(agg, 0, (size_t)N_NODES * L_DIM * 4, stream);
        edge_msg_kernel<<<NBLK, 128, 0, stream>>>(eidx, out16, h1, Bsw, agg);
        node_update_kernel<<<N_NODES, 64, 0, stream>>>(agg, invdeg, croot, cbias,
                                                       wih, whh, bih, bhh,
                                                       out32, out16, hbuf);
    }
    final_kernel<<<N_NODES, 64, 0, stream>>>(hbuf, l1w, l1b, l2w, l2b, yout);
}
